// Attention_7825430413968
// MI455X (gfx1250) — matrix operand order repaired and checked by reading, compiled, not run on hardware
//
#include <hip/hip_runtime.h>
#include <hip/hip_bf16.h>

// ---------------------------------------------------------------------------
// Problem constants (match reference)
// ---------------------------------------------------------------------------
#define BB 8
#define SS 2048
#define TT 2048
#define DD 1024

typedef __bf16 bf16_t;
typedef __attribute__((ext_vector_type(16))) __bf16 v16bf;
typedef __attribute__((ext_vector_type(8)))  __bf16 v8bf;
typedef __attribute__((ext_vector_type(8)))  float  v8f;

// ---------------------------------------------------------------------------
// WMMA helpers (CDNA5 wave32, v_wmma_f32_16x16x32_bf16)
// ---------------------------------------------------------------------------
static __device__ __forceinline__ v16bf cat8(v8bf lo, v8bf hi) {
    return __builtin_shufflevector(lo, hi, 0,1,2,3,4,5,6,7,8,9,10,11,12,13,14,15);
}

static __device__ __forceinline__ v8f wmma_bf16(v16bf a, v16bf b, v8f c) {
    // 8 args: (neg_a, A, neg_b, B, c_mod, C, reuse_a, reuse_b)
    return __builtin_amdgcn_wmma_f32_16x16x32_bf16(false, a, false, b, (short)0, c, false, false);
}

// A-operand 16x32 bf16 tile, row-major source pointed at tile origin.
// Lane L: row = L&15; K chunks {8h..8h+7} and {16+8h..23+8h}, h = L>=16.
static __device__ __forceinline__ v16bf load_A_bf(const bf16_t* __restrict__ tile,
                                                  int ld, unsigned lane) {
    const int m  = lane & 15;
    const int h  = (lane >> 4) & 1;
    const bf16_t* p = tile + (size_t)m * ld + h * 8;
    v8bf a0 = *(const v8bf*)(p);
    v8bf a1 = *(const v8bf*)(p + 16);
    return cat8(a0, a1);
}

// Same A layout but fp32 source, converted to bf16 on the fly.
static __device__ __forceinline__ v16bf load_A_f32(const float* __restrict__ tile,
                                                   int ld, unsigned lane) {
    const int m = lane & 15;
    const int h = (lane >> 4) & 1;
    const float* p = tile + (size_t)m * ld + h * 8;
    v8f f0 = *(const v8f*)(p);
    v8f f1 = *(const v8f*)(p + 16);
    v16bf a;
#pragma unroll
    for (int i = 0; i < 8; ++i) {
        a[i]     = (bf16_t)f0[i];
        a[i + 8] = (bf16_t)f1[i];
    }
    return a;
}

// B-operand 32x16: B[k][n] comes from memory row n (contiguous in k).
// Lane L: n = L&15; K chunks {8h..8h+7} and {16+8h..23+8h}.
static __device__ __forceinline__ v16bf load_B_bf(const bf16_t* __restrict__ tile,
                                                  int ld, unsigned lane) {
    const int n = lane & 15;
    const int h = (lane >> 4) & 1;
    const bf16_t* p = tile + (size_t)n * ld + h * 8;
    return cat8(*(const v8bf*)(p), *(const v8bf*)(p + 16));
}

// D/C tile (f32, 8 VGPRs): VGPR r, lane L -> (row r + 8*(L>=16), col L&15)
static __device__ __forceinline__ void store_D_f32(float* __restrict__ tile, int ld,
                                                   v8f c, unsigned lane) {
    const int n = lane & 15;
    const int h = (lane >> 4) & 1;
#pragma unroll
    for (int r = 0; r < 8; ++r)
        tile[(size_t)(r + 8 * h) * ld + n] = c[r];
}

static __device__ __forceinline__ void store_D_bf16(bf16_t* __restrict__ tile, int ld,
                                                    v8f c, unsigned lane) {
    const int n = lane & 15;
    const int h = (lane >> 4) & 1;
#pragma unroll
    for (int r = 0; r < 8; ++r)
        tile[(size_t)(r + 8 * h) * ld + n] = (bf16_t)c[r];
}

// ---------------------------------------------------------------------------
// Staging kernels
// ---------------------------------------------------------------------------
__global__ void k_f32_to_bf16(const float* __restrict__ src, bf16_t* __restrict__ dst,
                              int n) {
    int i      = blockIdx.x * blockDim.x + threadIdx.x;
    int stride = gridDim.x * blockDim.x;
    for (; i < n; i += stride) dst[i] = (bf16_t)src[i];
}

// Transpose src[R][C] (fp32) -> dst[C][R] (bf16), batched over blockIdx.z.
__global__ void k_transpose_bf16(const float* __restrict__ src, bf16_t* __restrict__ dst,
                                 int R, int C, long sbatch, long dbatch) {
    __shared__ bf16_t t[32][33];
    src += (long)blockIdx.z * sbatch;
    dst += (long)blockIdx.z * dbatch;
    const int x0 = blockIdx.x * 32;  // source column base
    const int y0 = blockIdx.y * 32;  // source row base
#pragma unroll
    for (int j = threadIdx.y; j < 32; j += 8)
        t[j][threadIdx.x] = (bf16_t)src[(size_t)(y0 + j) * C + (x0 + threadIdx.x)];
    __syncthreads();
#pragma unroll
    for (int j = threadIdx.y; j < 32; j += 8)
        dst[(size_t)(x0 + j) * R + (y0 + threadIdx.x)] = t[threadIdx.x][j];
}

// ---------------------------------------------------------------------------
// 32x32-per-wave micro-kernel: A rows at Abase (ldA), B rows at Bbase (ldB),
// K reduction length kk. acc laid out [mi][ni].
// ---------------------------------------------------------------------------
struct Acc2x2 { v8f c00, c01, c10, c11; };

static __device__ __forceinline__ void mk_gemm32_bf(const bf16_t* __restrict__ Abase, int ldA,
                                                    const bf16_t* __restrict__ Bbase, int ldB,
                                                    int kk, unsigned lane, Acc2x2& acc) {
    for (int k0 = 0; k0 < kk; k0 += 32) {
        __builtin_prefetch(Abase + k0 + 128, 0, 1);
        __builtin_prefetch(Bbase + k0 + 128, 0, 1);
        v16bf a0 = load_A_bf(Abase + k0, ldA, lane);
        v16bf a1 = load_A_bf(Abase + (size_t)16 * ldA + k0, ldA, lane);
        v16bf b0 = load_B_bf(Bbase + k0, ldB, lane);
        v16bf b1 = load_B_bf(Bbase + (size_t)16 * ldB + k0, ldB, lane);
        acc.c00 = wmma_bf16(a0, b0, acc.c00);
        acc.c01 = wmma_bf16(a0, b1, acc.c01);
        acc.c10 = wmma_bf16(a1, b0, acc.c10);
        acc.c11 = wmma_bf16(a1, b1, acc.c11);
    }
}

static __device__ __forceinline__ void mk_gemm32_f32A(const float* __restrict__ Abase, int ldA,
                                                      const bf16_t* __restrict__ Bbase, int ldB,
                                                      int kk, unsigned lane, Acc2x2& acc) {
    for (int k0 = 0; k0 < kk; k0 += 32) {
        __builtin_prefetch(Abase + k0 + 128, 0, 1);
        __builtin_prefetch(Bbase + k0 + 128, 0, 1);
        v16bf a0 = load_A_f32(Abase + k0, ldA, lane);
        v16bf a1 = load_A_f32(Abase + (size_t)16 * ldA + k0, ldA, lane);
        v16bf b0 = load_B_bf(Bbase + k0, ldB, lane);
        v16bf b1 = load_B_bf(Bbase + (size_t)16 * ldB + k0, ldB, lane);
        acc.c00 = wmma_bf16(a0, b0, acc.c00);
        acc.c01 = wmma_bf16(a0, b1, acc.c01);
        acc.c10 = wmma_bf16(a1, b0, acc.c10);
        acc.c11 = wmma_bf16(a1, b1, acc.c11);
    }
}

// ---------------------------------------------------------------------------
// GEMM 1: dw = dec @ W   ([B*T, D] x [D, D] -> bf16 [B*T, D])
// A = dec_bf rows, B = wT rows (wT[n][k] = W[k][n]). 32x32 per wave.
// ---------------------------------------------------------------------------
__global__ void k_dw_gemm(const bf16_t* __restrict__ dec, const bf16_t* __restrict__ wT,
                          bf16_t* __restrict__ dw) {
    const unsigned lane = threadIdx.x & 31u;
    const int g   = blockIdx.x * 8 + (threadIdx.x >> 5);
    const int tn  = g & (DD / 32 - 1);             // 32 col super-tiles
    const int tm  = g >> 5;                        // 512 row super-tiles
    const bf16_t* Abase = dec + (size_t)tm * 32 * DD;
    const bf16_t* Bbase = wT  + (size_t)tn * 32 * DD;
    Acc2x2 acc = {};
    mk_gemm32_bf(Abase, DD, Bbase, DD, DD, lane, acc);
    bf16_t* Dp = dw + (size_t)tm * 32 * DD + tn * 32;
    store_D_bf16(Dp,                        DD, acc.c00, lane);
    store_D_bf16(Dp + 16,                   DD, acc.c01, lane);
    store_D_bf16(Dp + (size_t)16 * DD,      DD, acc.c10, lane);
    store_D_bf16(Dp + (size_t)16 * DD + 16, DD, acc.c11, lane);
}

// ---------------------------------------------------------------------------
// GEMM 2: affine = dw @ enc^T  per batch  ([T,D] x [D,S] -> fp32 [T,S])
// B[k][n] = enc[s0+n][k]  -> enc_bf rows directly. Raw affine -> score region.
// ---------------------------------------------------------------------------
__global__ void k_affine_gemm(const bf16_t* __restrict__ dw, const bf16_t* __restrict__ enc,
                              float* __restrict__ score) {
    const unsigned lane = threadIdx.x & 31u;
    const int g     = blockIdx.x * 8 + (threadIdx.x >> 5);
    const int per_b = (TT / 32) * (SS / 32);       // 4096
    const int b     = g / per_b;
    const int rem   = g - b * per_b;
    const int tn    = rem & (SS / 32 - 1);
    const int tm    = rem >> 6;
    const bf16_t* Abase = dw  + (size_t)b * TT * DD + (size_t)tm * 32 * DD;
    const bf16_t* Bbase = enc + (size_t)b * SS * DD + (size_t)tn * 32 * DD;
    Acc2x2 acc = {};
    mk_gemm32_bf(Abase, DD, Bbase, DD, DD, lane, acc);
    float* Dp = score + (size_t)b * TT * SS + (size_t)tm * 32 * SS + tn * 32;
    store_D_f32(Dp,                        SS, acc.c00, lane);
    store_D_f32(Dp + 16,                   SS, acc.c01, lane);
    store_D_f32(Dp + (size_t)16 * SS,      SS, acc.c10, lane);
    store_D_f32(Dp + (size_t)16 * SS + 16, SS, acc.c11, lane);
}

// ---------------------------------------------------------------------------
// In-place row log-softmax over score [B*T rows, S cols].
// ---------------------------------------------------------------------------
__global__ void k_logsoftmax(float* __restrict__ score) {
    __shared__ float red[256];
    float* p = score + (size_t)blockIdx.x * SS;
    float v[SS / 256];
    float m = -1e30f;
#pragma unroll
    for (int i = 0; i < SS / 256; ++i) {
        v[i] = p[threadIdx.x + i * 256];
        m = fmaxf(m, v[i]);
    }
    red[threadIdx.x] = m;
    __syncthreads();
    for (int off = 128; off > 0; off >>= 1) {
        if (threadIdx.x < off)
            red[threadIdx.x] = fmaxf(red[threadIdx.x], red[threadIdx.x + off]);
        __syncthreads();
    }
    m = red[0];
    __syncthreads();
    float s = 0.f;
#pragma unroll
    for (int i = 0; i < SS / 256; ++i) s += __expf(v[i] - m);
    red[threadIdx.x] = s;
    __syncthreads();
    for (int off = 128; off > 0; off >>= 1) {
        if (threadIdx.x < off) red[threadIdx.x] += red[threadIdx.x + off];
        __syncthreads();
    }
    const float L = m + __logf(red[0]);
#pragma unroll
    for (int i = 0; i < SS / 256; ++i) p[threadIdx.x + i * 256] = v[i] - L;
}

// ---------------------------------------------------------------------------
// GEMM 3: ctx = score @ enc   per batch ([T,S] x [S,D] -> bf16 [T,D])
// A = score rows (fp32 -> bf16 on the fly), B[k][n] = enc[k][n] = encT[n][k].
// ---------------------------------------------------------------------------
__global__ void k_ctx_gemm(const float* __restrict__ score, const bf16_t* __restrict__ encT,
                           bf16_t* __restrict__ ctx) {
    const unsigned lane = threadIdx.x & 31u;
    const int g     = blockIdx.x * 8 + (threadIdx.x >> 5);
    const int per_b = (TT / 32) * (DD / 32);       // 2048
    const int b     = g / per_b;
    const int rem   = g - b * per_b;
    const int tn    = rem & (DD / 32 - 1);
    const int tm    = rem >> 5;
    const float*  Abase = score + (size_t)b * TT * SS + (size_t)tm * 32 * SS;
    const bf16_t* Bbase = encT  + (size_t)b * DD * SS + (size_t)tn * 32 * SS;
    Acc2x2 acc = {};
    mk_gemm32_f32A(Abase, SS, Bbase, SS, SS, lane, acc);
    bf16_t* Dp = ctx + (size_t)b * TT * DD + (size_t)tm * 32 * DD + tn * 32;
    store_D_bf16(Dp,                        DD, acc.c00, lane);
    store_D_bf16(Dp + 16,                   DD, acc.c01, lane);
    store_D_bf16(Dp + (size_t)16 * DD,      DD, acc.c10, lane);
    store_D_bf16(Dp + (size_t)16 * DD + 16, DD, acc.c11, lane);
}

// ---------------------------------------------------------------------------
// GEMM 4: out = ctx @ Pw[0:D] + dec @ Pw[D:2D] + bias
// pwT[n][k] = proj_w[k][n], pwT is [D][2D]. Rows of ctx/dec are A-operands.
// ---------------------------------------------------------------------------
__global__ void k_out_gemm(const bf16_t* __restrict__ ctx, const bf16_t* __restrict__ dec,
                           const bf16_t* __restrict__ pwT, const float* __restrict__ bias,
                           float* __restrict__ out) {
    const unsigned lane = threadIdx.x & 31u;
    const int g  = blockIdx.x * 8 + (threadIdx.x >> 5);
    const int tn = g & (DD / 32 - 1);
    const int tm = g >> 5;
    const bf16_t* A0 = ctx + (size_t)tm * 32 * DD;
    const bf16_t* A1 = dec + (size_t)tm * 32 * DD;
    const bf16_t* Bb = pwT + (size_t)tn * 32 * (2 * DD);
    const float bv0 = bias[tn * 32 + (lane & 15)];
    const float bv1 = bias[tn * 32 + 16 + (lane & 15)];
    Acc2x2 acc;
#pragma unroll
    for (int r = 0; r < 8; ++r) {
        acc.c00[r] = bv0; acc.c10[r] = bv0;
        acc.c01[r] = bv1; acc.c11[r] = bv1;
    }
    mk_gemm32_bf(A0, DD, Bb,      2 * DD, DD, lane, acc);   // ctx @ Pw[0:D]
    mk_gemm32_bf(A1, DD, Bb + DD, 2 * DD, DD, lane, acc);   // dec @ Pw[D:2D]
    float* Dp = out + (size_t)tm * 32 * DD + tn * 32;
    store_D_f32(Dp,                        DD, acc.c00, lane);
    store_D_f32(Dp + 16,                   DD, acc.c01, lane);
    store_D_f32(Dp + (size_t)16 * DD,      DD, acc.c10, lane);
    store_D_f32(Dp + (size_t)16 * DD + 16, DD, acc.c11, lane);
}

// ---------------------------------------------------------------------------
// Launcher
// ---------------------------------------------------------------------------
extern "C" void kernel_launch(void* const* d_in, const int* in_sizes, int n_in,
                              void* d_out, int out_size, void* d_ws, size_t ws_size,
                              hipStream_t stream) {
    const float* enc    = (const float*)d_in[0];   // [B,S,D]
    const float* dec    = (const float*)d_in[1];   // [B,T,D]
    const float* weight = (const float*)d_in[2];   // [D,D]
    const float* proj_w = (const float*)d_in[3];   // [2D,D]
    const float* proj_b = (const float*)d_in[4];   // [D]

    float* out   = (float*)d_out;                          // [B,T,D]
    float* score = out + (size_t)BB * TT * DD;             // [B,T,S]

    const size_t N_BTD = (size_t)BB * TT * DD;             // 16,777,216
    const size_t N_BSD = (size_t)BB * SS * DD;

    char* ws = (char*)d_ws;
    size_t off = 0;
    auto take = [&](size_t bytes) { void* p = ws + off; off = (off + bytes + 255) & ~(size_t)255; return p; };
    bf16_t* dec_bf  = (bf16_t*)take(N_BTD * sizeof(bf16_t));
    bf16_t* enc_bf  = (bf16_t*)take(N_BSD * sizeof(bf16_t));
    bf16_t* encT_bf = (bf16_t*)take(N_BSD * sizeof(bf16_t));
    bf16_t* wT_bf   = (bf16_t*)take((size_t)DD * DD * sizeof(bf16_t));
    bf16_t* pwT_bf  = (bf16_t*)take((size_t)DD * 2 * DD * sizeof(bf16_t));
    bf16_t* dw_bf   = (bf16_t*)take(N_BTD * sizeof(bf16_t));
    bf16_t* ctx_bf  = (bf16_t*)take(N_BTD * sizeof(bf16_t));
    (void)ws_size; (void)n_in; (void)in_sizes; (void)out_size;

    // Stage bf16 copies / transposes
    k_f32_to_bf16<<<2048, 256, 0, stream>>>(dec, dec_bf, (int)N_BTD);
    k_f32_to_bf16<<<2048, 256, 0, stream>>>(enc, enc_bf, (int)N_BSD);
    {   // W [D][D] -> wT [D][D]
        dim3 g(DD / 32, DD / 32, 1), b(32, 8, 1);
        k_transpose_bf16<<<g, b, 0, stream>>>(weight, wT_bf, DD, DD, 0, 0);
    }
    {   // proj_w [2D][D] -> pwT [D][2D]
        dim3 g(DD / 32, (2 * DD) / 32, 1), b(32, 8, 1);
        k_transpose_bf16<<<g, b, 0, stream>>>(proj_w, pwT_bf, 2 * DD, DD, 0, 0);
    }
    {   // enc [b][S][D] -> encT [b][D][S]
        dim3 g(DD / 32, SS / 32, BB), b(32, 8, 1);
        k_transpose_bf16<<<g, b, 0, stream>>>(enc, encT_bf, SS, DD,
                                              (long)SS * DD, (long)DD * SS);
    }

    // GEMM 1: dw = dec @ W         (512*32 = 16384 wave-tiles / 8 waves)
    k_dw_gemm<<<2048, 256, 0, stream>>>(dec_bf, wT_bf, dw_bf);
    // GEMM 2: affine -> score region (raw)   (8*64*64 = 32768 wave-tiles)
    k_affine_gemm<<<4096, 256, 0, stream>>>(dw_bf, enc_bf, score);
    // Row-wise log-softmax in place
    k_logsoftmax<<<BB * TT, 256, 0, stream>>>(score);
    // GEMM 3: ctx = score @ enc    (8*64*32 = 16384 wave-tiles)
    k_ctx_gemm<<<2048, 256, 0, stream>>>(score, encT_bf, ctx_bf);
    // GEMM 4: out = [ctx,dec] @ proj_w + b   (512*32 wave-tiles)
    k_out_gemm<<<2048, 256, 0, stream>>>(ctx_bf, dec_bf, pwT_bf, proj_b, out);
}
